// LlamaAttention_KIVI_14353780703725
// MI455X (gfx1250) — matrix operand order repaired and checked by reading, not compiled, not run on hardware
//
#include <hip/hip_runtime.h>

// ---------------------------------------------------------------------------
// Llama attention block for MI455X (gfx1250), bf16 WMMA everywhere.
// Shapes: B=1, S=2048, H=4096, NH=32, KVH=8, HD=128, GROUPS=4.
// LDS staging uses GLOBAL_LOAD_ASYNC_TO_LDS_B128 (ASYNCcnt) when available.
// ---------------------------------------------------------------------------

typedef unsigned short u16;
typedef unsigned int   u32;

typedef __bf16 bf16;
typedef bf16  v16bf __attribute__((ext_vector_type(16)));
typedef float v8f   __attribute__((ext_vector_type(8)));
typedef int   v4i   __attribute__((ext_vector_type(4)));

#define AS1 __attribute__((address_space(1)))
#define AS3 __attribute__((address_space(3)))

#if defined(__has_builtin)
#  if __has_builtin(__builtin_amdgcn_global_load_async_to_lds_b128)
#    define HAVE_ASYNC_LDS 1
#  else
#    define HAVE_ASYNC_LDS 0
#  endif
#  if __has_builtin(__builtin_amdgcn_s_wait_asynccnt)
#    define HAVE_WAIT_ASYNC 1
#  else
#    define HAVE_WAIT_ASYNC 0
#  endif
#else
#  define HAVE_ASYNC_LDS 0
#  define HAVE_WAIT_ASYNC 0
#endif

union Frag {
  v16bf v;
  uint4 q[2];
};

__device__ __forceinline__ u16 f2bf(float f) {
  u32 u = __float_as_uint(f);
  u += 0x7FFFu + ((u >> 16) & 1u);   // round-to-nearest-even
  return (u16)(u >> 16);
}
__device__ __forceinline__ float bf2f(u16 h) {
  return __uint_as_float(((u32)h) << 16);
}
__device__ __forceinline__ v8f wmma_bf16(v16bf a, v16bf b, v8f c) {
  // 8 args: (neg_a, A, neg_b, B, c_mod, C, reuse_a, reuse_b)
  return __builtin_amdgcn_wmma_f32_16x16x32_bf16(false, a, false, b,
                                                 (short)0, c, false, false);
}

// 16-byte global -> LDS copy. Async path on gfx1250 toolchains that expose
// the builtin (GLOBAL_LOAD_ASYNC_TO_LDS_B128, tracked by ASYNCcnt),
// plain load + ds_store otherwise. Builtin parameter type is int4* per the
// toolchain's declaration (b128 payload).
__device__ __forceinline__ void copy16_g2l(void* lds_dst, const void* gsrc) {
#if HAVE_ASYNC_LDS
  __builtin_amdgcn_global_load_async_to_lds_b128(
      (AS1 v4i*)(unsigned long long)gsrc,
      (AS3 v4i*)(unsigned)(unsigned long long)lds_dst,
      0, 0);
#else
  *(uint4*)lds_dst = *(const uint4*)gsrc;
#endif
}

// Fence between issuing staging copies and consuming the LDS tile.
__device__ __forceinline__ void stage_fence() {
#if HAVE_ASYNC_LDS
#  if HAVE_WAIT_ASYNC
  __builtin_amdgcn_s_wait_asynccnt(0);
#  else
  asm volatile("s_wait_asynccnt 0" ::: "memory");
#  endif
#endif
  __syncthreads();
}

#define S_LEN   2048
#define HID     4096
#define NHEADS  32
#define KVHEADS 8
#define HDIM    128

// ---------------------------------------------------------------------------
// fp32 -> bf16 conversion
// ---------------------------------------------------------------------------
__global__ void cvt_f32_bf16(const float* __restrict__ in, u16* __restrict__ out,
                             size_t n) {
  size_t i = (size_t)blockIdx.x * blockDim.x + threadIdx.x;
  if (i < n) out[i] = f2bf(in[i]);
}

// ---------------------------------------------------------------------------
// Generic bf16 WMMA GEMM:  C[M,N] = A[M,K] * W[N,K]^T
//   mode 0: C fp32 row-major (ldc = N)
//   mode 1: C bf16 row-major (ldc = N)
//   mode 2: C bf16 transposed, C[n*ldc + m] (ldc = M)   (used for V^T)
// Block: 256 threads (8 waves). Block tile 128x128, wave tile 32x64.
// ---------------------------------------------------------------------------
__global__ __launch_bounds__(256)
void gemm_bf16_wmma(const u16* __restrict__ A, const u16* __restrict__ W,
                    void* __restrict__ Cout, int M, int N, int K,
                    int mode, int ldc) {
  __shared__ __align__(16) u16 As[128][32];   // A tile: 128 rows x 32 K (8 KB)

  const int t    = threadIdx.x;
  const int lane = t & 31;
  const int wid  = t >> 5;
  const int half = lane >> 4;       // 0: lanes 0-15, 1: lanes 16-31
  const int l16  = lane & 15;
  const int wm   = wid & 3;         // 4 m-strips of 32 rows
  const int wn   = wid >> 2;        // 2 n-strips of 64 cols
  const int bm   = blockIdx.y * 128;
  const int bn   = blockIdx.x * 128;

  v8f acc[2][4];
#pragma unroll
  for (int mt = 0; mt < 2; ++mt)
#pragma unroll
    for (int nt = 0; nt < 4; ++nt) acc[mt][nt] = {};

  const int arow = t >> 1;          // 0..127
  const int acol = (t & 1) * 16;    // 0 or 16

  for (int kk = 0; kk < K; kk += 32) {
    __syncthreads();
    { // cooperative A-tile staging: each thread 32 bytes (2x b128)
      const u16* src = A + (size_t)(bm + arow) * K + kk + acol;
      copy16_g2l(&As[arow][acol], src);
      copy16_g2l(&As[arow][acol + 8], src + 8);
    }
    stage_fence();

    // A fragments (16x32 bf16): contiguous K run of 16 elems at 16*half
    Frag a[2];
#pragma unroll
    for (int mt = 0; mt < 2; ++mt) {
      const u16* p = &As[wm * 32 + mt * 16 + l16][16 * half];
      a[mt].q[0] = *(const uint4*)p;
      a[mt].q[1] = *(const uint4*)(p + 8);
    }

#pragma unroll
    for (int nt = 0; nt < 4; ++nt) {
      // B fragment (32x16 bf16): contiguous K run of 16 elems at 16*half
      const u16* wp = W + (size_t)(bn + wn * 64 + nt * 16 + l16) * K
                        + kk + 16 * half;
      Frag b;
      b.q[0] = *(const uint4*)wp;
      b.q[1] = *(const uint4*)(wp + 8);
      if (kk + 32 < K) __builtin_prefetch(wp + 32, 0, 3);  // global_prefetch_b8
#pragma unroll
      for (int mt = 0; mt < 2; ++mt)
        acc[mt][nt] = wmma_bf16(a[mt].v, b.v, acc[mt][nt]);
    }
  }

  // D layout: lane holds col n = l16; VGPR r holds row r + 8*half
#pragma unroll
  for (int mt = 0; mt < 2; ++mt)
#pragma unroll
    for (int nt = 0; nt < 4; ++nt)
#pragma unroll
      for (int r = 0; r < 8; ++r) {
        int m = bm + wm * 32 + mt * 16 + r + 8 * half;
        int n = bn + wn * 64 + nt * 16 + l16;
        float v = acc[mt][nt][r];
        if (mode == 0)      ((float*)Cout)[(size_t)m * ldc + n] = v;
        else if (mode == 1) ((u16*)Cout)[(size_t)m * ldc + n] = f2bf(v);
        else                ((u16*)Cout)[(size_t)n * ldc + m] = f2bf(v);
      }
}

// ---------------------------------------------------------------------------
// In-place RoPE on bf16 [S, nheads*128]; thread handles one (s, h, j<64) pair.
// ---------------------------------------------------------------------------
__global__ void rope_bf16(u16* __restrict__ qk, const int* __restrict__ pos_ids,
                          int nheads) {
  size_t idx = (size_t)blockIdx.x * blockDim.x + threadIdx.x;
  size_t total = (size_t)S_LEN * nheads * 64;
  if (idx >= total) return;
  int j = (int)(idx & 63);
  size_t tmp = idx >> 6;
  int h = (int)(tmp % nheads);
  int s = (int)(tmp / nheads);

  float pos = (float)pos_ids[s];
  // inv_freq = theta^(-2j/128); log(10000) = 9.2103403719...
  float inv = __expf(-(float)(2 * j) * (9.210340371976184f / 128.0f));
  float ang = pos * inv;
  float sn, cs;
  __sincosf(ang, &sn, &cs);

  size_t base = (size_t)s * ((size_t)nheads * HDIM) + (size_t)h * HDIM;
  float x0 = bf2f(qk[base + j]);
  float x1 = bf2f(qk[base + j + 64]);
  qk[base + j]      = f2bf(x0 * cs - x1 * sn);
  qk[base + j + 64] = f2bf(x1 * cs + x0 * sn);
}

// ---------------------------------------------------------------------------
// Flash attention (causal, GQA). Grid: (NHEADS, S/128). Block: 256 (8 waves).
// Each wave owns 16 q-rows. K processed in 64-wide key blocks staged in LDS.
//   Qb : [S, 4096] bf16 (post-RoPE)    Kb : [S, 1024] bf16 (post-RoPE)
//   Vt : [1024, S] bf16 (transposed)   Ab : [S, 4096] bf16 attention output
// ---------------------------------------------------------------------------
__global__ __launch_bounds__(256)
void attn_flash_wmma(const u16* __restrict__ Qb, const u16* __restrict__ Kb,
                     const u16* __restrict__ Vt, u16* __restrict__ Ab) {
  __shared__ __align__(16) u16 Ks[64][128];    // key block  [s][d]  16 KB
  __shared__ __align__(16) u16 Vs[128][64];    // V^T block  [d][s]  16 KB
  __shared__ __align__(16) u16 Ps[8][16][64];  // per-wave P strips  16 KB

  const int h   = blockIdx.x;          // query head
  const int qb  = blockIdx.y;          // q block (128 rows)
  const int kvh = h >> 2;              // GQA: kv head = h / GROUPS
  const int t    = threadIdx.x;
  const int lane = t & 31;
  const int wid  = t >> 5;
  const int half = lane >> 4;
  const int l16  = lane & 15;

  // Preload this wave's Q fragments (16 rows x 128 d = 4 k-steps) into regs.
  Frag qf[4];
  const u16* qrow = Qb + (size_t)(qb * 128 + wid * 16 + l16) * HID + h * HDIM;
#pragma unroll
  for (int ks = 0; ks < 4; ++ks) {
    const u16* p = qrow + ks * 32 + 16 * half;
    qf[ks].q[0] = *(const uint4*)p;
    qf[ks].q[1] = *(const uint4*)(p + 8);
  }

  v8f acc[8];
#pragma unroll
  for (int dt = 0; dt < 8; ++dt) acc[dt] = {};
  v8f mrow, lrow;
#pragma unroll
  for (int r = 0; r < 8; ++r) { mrow[r] = -3.0e38f; lrow[r] = 0.0f; }

  const float scale = 0.08838834764831845f;  // 1/sqrt(128)
  const int nkb = 2 * qb + 2;                // key blocks (causal)

  for (int kb = 0; kb < nkb; ++kb) {
    __syncthreads();
    // Stage K block [64 x 128] and V^T block [128 x 64] into LDS (async).
    for (int i = t; i < 1024; i += 256) {
      int row = i >> 4, col = (i & 15) * 8;
      copy16_g2l(&Ks[row][col],
                 Kb + (size_t)(kb * 64 + row) * (KVHEADS * HDIM)
                    + kvh * HDIM + col);
    }
    for (int i = t; i < 1024; i += 256) {
      int row = i >> 3, col = (i & 7) * 8;
      copy16_g2l(&Vs[row][col],
                 Vt + (size_t)(kvh * HDIM + row) * S_LEN + kb * 64 + col);
    }
    stage_fence();

    // --- scores: 16 q-rows x 64 keys via WMMA (K-dim = head dim 128) ---
    v8f sc[4];
#pragma unroll
    for (int nt = 0; nt < 4; ++nt) {
      v8f s = {};
#pragma unroll
      for (int ks = 0; ks < 4; ++ks) {
        const u16* p = &Ks[nt * 16 + l16][ks * 32 + 16 * half];
        Frag b;
        b.q[0] = *(const uint4*)p;
        b.q[1] = *(const uint4*)(p + 8);
        s = wmma_bf16(qf[ks].v, b.v, s);
      }
      s = s * scale;
#pragma unroll
      for (int r = 0; r < 8; ++r) {   // causal mask
        int col = kb * 64 + nt * 16 + l16;
        int row = qb * 128 + wid * 16 + r + 8 * half;
        if (col > row) s[r] = -1.0e30f;
      }
      sc[nt] = s;
    }

    // --- online softmax (row = across 16 lanes of the half + 4 n-tiles) ---
    v8f alv;
#pragma unroll
    for (int r = 0; r < 8; ++r) {
      float mx = sc[0][r];
#pragma unroll
      for (int nt = 1; nt < 4; ++nt) mx = fmaxf(mx, sc[nt][r]);
#pragma unroll
      for (int m = 1; m < 16; m <<= 1)
        mx = fmaxf(mx, __shfl_xor(mx, m, 16));
      float mnew  = fmaxf(mrow[r], mx);
      float alpha = __expf(mrow[r] - mnew);
      float rs = 0.0f;
#pragma unroll
      for (int nt = 0; nt < 4; ++nt) {
        float pv = __expf(sc[nt][r] - mnew);
        sc[nt][r] = pv;
        rs += pv;
      }
#pragma unroll
      for (int m = 1; m < 16; m <<= 1)
        rs += __shfl_xor(rs, m, 16);
      lrow[r] = lrow[r] * alpha + rs;
      mrow[r] = mnew;
      alv[r]  = alpha;
    }
#pragma unroll
    for (int dt = 0; dt < 8; ++dt) acc[dt] *= alv;

    // --- re-layout P (D-frag -> A-frag) through this wave's LDS strip ---
#pragma unroll
    for (int nt = 0; nt < 4; ++nt)
#pragma unroll
      for (int r = 0; r < 8; ++r)
        Ps[wid][r + 8 * half][nt * 16 + l16] = f2bf(sc[nt][r]);
    // wave-private region: DS ops are in-order within a wave, no barrier.

    Frag ap[2];
#pragma unroll
    for (int k2 = 0; k2 < 2; ++k2) {
      const u16* p = &Ps[wid][l16][k2 * 32 + 16 * half];
      ap[k2].q[0] = *(const uint4*)p;
      ap[k2].q[1] = *(const uint4*)(p + 8);
    }

    // --- out += P @ V  (K-dim = 64 keys, N = 128 head dims) ---
#pragma unroll
    for (int dt = 0; dt < 8; ++dt)
#pragma unroll
      for (int k2 = 0; k2 < 2; ++k2) {
        const u16* p = &Vs[dt * 16 + l16][k2 * 32 + 16 * half];
        Frag b;
        b.q[0] = *(const uint4*)p;
        b.q[1] = *(const uint4*)(p + 8);
        acc[dt] = wmma_bf16(ap[k2].v, b.v, acc[dt]);
      }
  }

  // Epilogue: normalize and store bf16.
#pragma unroll
  for (int dt = 0; dt < 8; ++dt)
#pragma unroll
    for (int r = 0; r < 8; ++r) {
      int row = qb * 128 + wid * 16 + r + 8 * half;
      int col = dt * 16 + l16;
      Ab[(size_t)row * HID + h * HDIM + col] = f2bf(acc[dt][r] / lrow[r]);
    }
}

// ---------------------------------------------------------------------------
// Host-side orchestration
// ---------------------------------------------------------------------------
extern "C" void kernel_launch(void* const* d_in, const int* in_sizes, int n_in,
                              void* d_out, int out_size, void* d_ws, size_t ws_size,
                              hipStream_t stream) {
  (void)in_sizes; (void)n_in; (void)out_size; (void)ws_size;
  const float* X   = (const float*)d_in[0];   // [1, 2048, 4096]
  const float* Wq  = (const float*)d_in[1];   // [4096, 4096]
  const float* Wk  = (const float*)d_in[2];   // [1024, 4096]
  const float* Wv  = (const float*)d_in[3];   // [1024, 4096]
  const float* Wo  = (const float*)d_in[4];   // [4096, 4096]
  const int*   pos = (const int*)d_in[5];     // [1, 2048]

  char* ws = (char*)d_ws;
  size_t off = 0;
  auto carve = [&](size_t bytes) -> void* {
    void* p = ws + off;
    off += (bytes + 255) & ~(size_t)255;
    return p;
  };
  u16* Xb  = (u16*)carve((size_t)S_LEN * HID * 2);          // 16 MB
  u16* Wqb = (u16*)carve((size_t)HID * HID * 2);            // 32 MB
  u16* Wkb = (u16*)carve((size_t)KVHEADS * HDIM * HID * 2); //  8 MB
  u16* Wvb = (u16*)carve((size_t)KVHEADS * HDIM * HID * 2); //  8 MB
  u16* Wob = (u16*)carve((size_t)HID * HID * 2);            // 32 MB
  u16* Qb  = (u16*)carve((size_t)S_LEN * HID * 2);          // 16 MB
  u16* Kb  = (u16*)carve((size_t)S_LEN * KVHEADS * HDIM * 2); // 4 MB
  u16* Vt  = (u16*)carve((size_t)KVHEADS * HDIM * S_LEN * 2); // 4 MB
  u16* Ab  = Xb;  // hidden-state bf16 copy is dead after QKV projections

  // 1) fp32 -> bf16 conversions
  {
    size_t n;
    n = (size_t)S_LEN * HID;
    cvt_f32_bf16<<<dim3((n + 255) / 256), dim3(256), 0, stream>>>(X, Xb, n);
    n = (size_t)HID * HID;
    cvt_f32_bf16<<<dim3((n + 255) / 256), dim3(256), 0, stream>>>(Wq, Wqb, n);
    n = (size_t)KVHEADS * HDIM * HID;
    cvt_f32_bf16<<<dim3((n + 255) / 256), dim3(256), 0, stream>>>(Wk, Wkb, n);
    cvt_f32_bf16<<<dim3((n + 255) / 256), dim3(256), 0, stream>>>(Wv, Wvb, n);
    n = (size_t)HID * HID;
    cvt_f32_bf16<<<dim3((n + 255) / 256), dim3(256), 0, stream>>>(Wo, Wob, n);
  }

  // 2) QKV projections (bf16 WMMA GEMMs)
  gemm_bf16_wmma<<<dim3(HID / 128, S_LEN / 128), dim3(256), 0, stream>>>(
      Xb, Wqb, Qb, S_LEN, HID, HID, /*mode=*/1, /*ldc=*/HID);
  gemm_bf16_wmma<<<dim3((KVHEADS * HDIM) / 128, S_LEN / 128), dim3(256), 0, stream>>>(
      Xb, Wkb, Kb, S_LEN, KVHEADS * HDIM, HID, /*mode=*/1, /*ldc=*/KVHEADS * HDIM);
  gemm_bf16_wmma<<<dim3((KVHEADS * HDIM) / 128, S_LEN / 128), dim3(256), 0, stream>>>(
      Xb, Wvb, Vt, S_LEN, KVHEADS * HDIM, HID, /*mode=*/2, /*ldc=*/S_LEN);

  // 3) RoPE (in place on bf16 Q and K)
  {
    size_t nq = (size_t)S_LEN * NHEADS * 64;
    rope_bf16<<<dim3((nq + 255) / 256), dim3(256), 0, stream>>>(Qb, pos, NHEADS);
    size_t nk = (size_t)S_LEN * KVHEADS * 64;
    rope_bf16<<<dim3((nk + 255) / 256), dim3(256), 0, stream>>>(Kb, pos, KVHEADS);
  }

  // 4) causal GQA flash attention
  attn_flash_wmma<<<dim3(NHEADS, S_LEN / 128), dim3(256), 0, stream>>>(
      Qb, Kb, Vt, Ab);

  // 5) output projection -> fp32 d_out
  gemm_bf16_wmma<<<dim3(HID / 128, S_LEN / 128), dim3(256), 0, stream>>>(
      Ab, Wob, d_out, S_LEN, HID, HID, /*mode=*/0, /*ldc=*/HID);
}
